// CNN3DRNN_6665789243912
// MI455X (gfx1250) — matrix operand order repaired and checked by reading, not compiled, not run on hardware
//
#include <hip/hip_runtime.h>
#include <hip/hip_bf16.h>

typedef _Float16 half_t;
typedef __attribute__((ext_vector_type(16))) _Float16 v16h;
typedef __attribute__((ext_vector_type(8)))  float    v8f;
typedef __attribute__((ext_vector_type(4)))  float    f32x4;
typedef __attribute__((ext_vector_type(4)))  int      i32x4;

union V16H { v16h v; f32x4 f[2]; };

// CDNA5 async global->LDS path, guarded so the file compiles on any toolchain.
#if defined(__HIP_DEVICE_COMPILE__) && defined(__has_builtin)
#if __has_builtin(__builtin_amdgcn_global_load_async_to_lds_b128) && \
    __has_builtin(__builtin_amdgcn_s_wait_asynccnt)
#define CDNA5_ASYNC_LDS 1
#endif
#endif
#ifndef CDNA5_ASYNC_LDS
#define CDNA5_ASYNC_LDS 0
#endif

#if CDNA5_ASYNC_LDS
typedef __attribute__((address_space(1))) i32x4 g_i32x4;   // global (AS1)
typedef __attribute__((address_space(3))) i32x4 l_i32x4;   // LDS    (AS3)
#endif

// ---------------------------------------------------------------------------
// fp32 -> f16 weight conversion
// ---------------------------------------------------------------------------
__global__ __launch_bounds__(256)
void f32_to_f16(const float* __restrict__ s, half_t* __restrict__ d, int n) {
    int i = blockIdx.x * 256 + threadIdx.x;
    if (i < n) d[i] = (half_t)s[i];
}

// ---------------------------------------------------------------------------
// Layer 1: direct conv3d SAME, Cin=3 -> Cout=32, reads windows from raw input.
// Output: (26, 8, 56, 56, 32) f16, item n = b*13 + tw, frame d -> time tw+d.
// ---------------------------------------------------------------------------
__global__ __launch_bounds__(256)
void conv1_direct(const float* __restrict__ in, const float* __restrict__ k,
                  const float* __restrict__ bias, half_t* __restrict__ y) {
    const int o = blockIdx.x * 256 + threadIdx.x;
    const int total = 26 * 8 * 56 * 56 * 32;
    if (o >= total) return;
    const int cout = o & 31;
    int p = o >> 5;
    const int w = p % 56; p /= 56;
    const int h = p % 56; p /= 56;
    const int d = p & 7;  p >>= 3;
    const int n = p;
    const int b = n / 13, tw = n % 13;
    float acc = bias[cout];
    for (int dz = -1; dz <= 1; ++dz) {
        const int f = d + dz;
        if ((unsigned)f >= 8u) continue;
        const int t = tw + f;
        for (int dy = -1; dy <= 1; ++dy) {
            const int hs = h + dy;
            if ((unsigned)hs >= 56u) continue;
            for (int dx = -1; dx <= 1; ++dx) {
                const int ws2 = w + dx;
                if ((unsigned)ws2 >= 56u) continue;
                const float* ip = in + ((size_t)((b * 20 + t) * 56 + hs) * 56 + ws2) * 3;
                const int tap = (dz + 1) * 9 + (dy + 1) * 3 + (dx + 1);
                const float* kp = k + (size_t)tap * 3 * 32 + cout;
                acc += ip[0] * kp[0] + ip[1] * kp[32] + ip[2] * kp[64];
            }
        }
    }
    y[o] = (half_t)fmaxf(acc, 0.0f);
}

// ---------------------------------------------------------------------------
// Implicit-GEMM conv3d SAME (+bias, +relu) via V_WMMA_F32_16X16X32_F16.
// x: (26, 8, HH, WW, CIN) f16 NDHWC; wgt: [27*CIN][COUT] f16 (row-major K x N);
// y: (..., COUT) f16.  M = 26*8*HH*WW rows (multiple of 64 here).
// Block = 256 threads = 8 waves; tile = 64 rows x COUT cols.
// K-chunks of 32 stay inside one tap because CIN % 32 == 0, so the A gather is
// a contiguous 16B per thread -> staged with GLOBAL_LOAD_ASYNC_TO_LDS_B128
// (ASYNCcnt) when available; SAME-padding lanes zero-fill LDS directly.
// ---------------------------------------------------------------------------
template <int CIN, int COUT, int HH, int WW>
__global__ __launch_bounds__(256)
void conv_wmma(const half_t* __restrict__ x, const half_t* __restrict__ wgt,
               const float* __restrict__ bias, half_t* __restrict__ y, int M) {
    constexpr int K  = 27 * CIN;
    constexpr int NT = COUT / 32;   // 16-col WMMA tiles per wave (2 waves split N)

    __shared__ half_t lA[64 * 32];        // [row][k]       4 KB
    __shared__ half_t lB[COUT * 32];      // [col][k]  up to 8 KB (transposed)

    const int tid   = threadIdx.x;
    const int wid   = tid >> 5;
    const int lane  = tid & 31;
    const int wm    = wid & 3;            // M tile 0..3
    const int wn    = wid >> 2;           // N half 0..1
    const int ln16  = lane & 15;
    const int khalf = lane >> 4;
    const int rowBase = blockIdx.x * 64;

    v8f acc[NT];
    for (int j = 0; j < NT; ++j)
        for (int e = 0; e < 8; ++e) acc[j][e] = 0.0f;

    // per-thread A-load coordinates (4 threads per row, 8 halves each)
    const int arow = tid >> 2;
    const int aseg = tid & 3;
    const int gr = rowBase + arow;
    const bool rValid = gr < M;
    int n = 0, d = 0, h = 0, w = 0;
    if (rValid) {
        w = gr % WW; int t1 = gr / WW;        // compile-time divisors
        h = t1 % HH; int t2 = t1 / HH;
        d = t2 & 7;  n = t2 >> 3;
    }
    half_t* aDst = lA + arow * 32 + aseg * 8;   // 16B per thread, 16B aligned

    for (int k0 = 0; k0 < K; k0 += 32) {
        const int tap = k0 / CIN;
        const int cb  = k0 % CIN;
        const int dz = tap / 9 - 1, dy = (tap / 3) % 3 - 1, dx = tap % 3 - 1;

        // stage A tile (zero-fill SAME padding)
        {
            const int ds = d + dz, hs = h + dy, ws2 = w + dx;
            const bool vld = rValid && (unsigned)ds < 8u &&
                             (unsigned)hs < (unsigned)HH && (unsigned)ws2 < (unsigned)WW;
            const half_t* src =
                x + ((size_t)(((n * 8 + ds) * HH + hs) * WW + ws2)) * CIN + cb + aseg * 8;
#if CDNA5_ASYNC_LDS
            if (vld) {
                __builtin_amdgcn_global_load_async_to_lds_b128(
                    (g_i32x4*)src, (l_i32x4*)aDst, 0, 0);
            } else {
                f32x4 z = {0.f, 0.f, 0.f, 0.f};
                *(f32x4*)aDst = z;
            }
#else
            f32x4 val = {0.f, 0.f, 0.f, 0.f};
            if (vld) val = *(const f32x4*)src;
            *(f32x4*)aDst = val;
#endif
        }
        // stage B tile transposed: lB[col][k]
        for (int e = tid; e < 32 * COUT; e += 256) {
            const int kk = e / COUT, col = e % COUT;
            lB[col * 32 + kk] = wgt[(size_t)(k0 + kk) * COUT + col];
        }
        // prefetch next chunk's weight rows (wave 0, one cacheline per lane)
        if (k0 + 32 < K && wid == 0) {
            __builtin_prefetch(wgt + (size_t)(k0 + 32) * COUT + lane * 32, 0, 1);
        }
#if CDNA5_ASYNC_LDS
        __builtin_amdgcn_s_wait_asynccnt(0);
#endif
        __syncthreads();

        // A fragment (16x32 f16): two contiguous 16B LDS reads per lane
        V16H a;
        const half_t* ap = lA + (wm * 16 + ln16) * 32 + khalf * 8;
        a.f[0] = *(const f32x4*)(ap);
        a.f[1] = *(const f32x4*)(ap + 16);

#pragma unroll
        for (int j = 0; j < NT; ++j) {
            V16H bfr;
            const half_t* bp = lB + (wn * (COUT / 2) + j * 16 + ln16) * 32 + khalf * 8;
            bfr.f[0] = *(const f32x4*)(bp);
            bfr.f[1] = *(const f32x4*)(bp + 16);
            acc[j] = __builtin_amdgcn_wmma_f32_16x16x32_f16(
                false, a.v, false, bfr.v, (short)0, acc[j], false, false);
        }
        __syncthreads();
    }

    // epilogue: bias + relu + f16 store
#pragma unroll
    for (int j = 0; j < NT; ++j) {
        const int col = wn * (COUT / 2) + j * 16 + ln16;
        const float bs = bias[col];
#pragma unroll
        for (int r = 0; r < 8; ++r) {
            const int orow = rowBase + wm * 16 + khalf * 8 + r;
            if (orow < M) {
                float v = fmaxf(acc[j][r] + bs, 0.0f);
                y[(size_t)orow * COUT + col] = (half_t)v;
            }
        }
    }
}

// ---------------------------------------------------------------------------
// MaxPool (1,2,2) valid, NDHWC f16.
// ---------------------------------------------------------------------------
__global__ __launch_bounds__(256)
void maxpool2x2(const half_t* __restrict__ x, half_t* __restrict__ y,
                int Hi, int Wi, int C, int total) {
    int o = blockIdx.x * 256 + threadIdx.x;
    if (o >= total) return;
    const int c = o % C; int p = o / C;
    const int Wo = Wi >> 1, Ho = Hi >> 1;
    const int w = p % Wo; p /= Wo;
    const int h = p % Ho; p /= Ho;          // p now = n*8+d
    const size_t base = (((size_t)p * Hi + 2 * h) * Wi + 2 * w) * C + c;
    const float a  = (float)x[base];
    const float b2 = (float)x[base + C];
    const float c2 = (float)x[base + (size_t)Wi * C];
    const float d2 = (float)x[base + (size_t)Wi * C + C];
    y[o] = (half_t)fmaxf(fmaxf(a, b2), fmaxf(c2, d2));
}

// ---------------------------------------------------------------------------
// Global average pool over (8,14,14): (26,8,14,14,128) f16 -> (26,128) f32.
// ---------------------------------------------------------------------------
__global__ __launch_bounds__(256)
void gap_kernel(const half_t* __restrict__ x, float* __restrict__ feats) {
    const int i = blockIdx.x * 256 + threadIdx.x;
    if (i >= 26 * 128) return;
    const int c = i & 127, n = i >> 7;
    const half_t* p = x + (size_t)n * 8 * 14 * 14 * 128 + c;
    float s = 0.f;
    for (int q = 0; q < 8 * 14 * 14; ++q) s += (float)p[(size_t)q * 128];
    feats[i] = s * (1.0f / 1568.0f);
}

// ---------------------------------------------------------------------------
// GRU input projection: xp[r][j] = b0[j] + sum_q x[r][q]*wx[q][j], j<192.
// ---------------------------------------------------------------------------
__global__ __launch_bounds__(256)
void gru_xp(const float* __restrict__ x, const float* __restrict__ wx,
            const float* __restrict__ b0, float* __restrict__ xp,
            int rows, int fin) {
    const int i = blockIdx.x * 256 + threadIdx.x;
    if (i >= rows * 192) return;
    const int j = i % 192, r = i / 192;
    float s = b0[j];
    const float* xr = x + (size_t)r * fin;
    for (int q = 0; q < fin; ++q) s += xr[q] * wx[(size_t)q * 192 + j];
    xp[i] = s;
}

// ---------------------------------------------------------------------------
// GRU sequential scan (reset_after=True). 1 block, 128 threads (2 batch x 64u).
// ---------------------------------------------------------------------------
__global__ __launch_bounds__(128)
void gru_scan(const float* __restrict__ xp, const float* __restrict__ wh,
              const float* __restrict__ b1, float* __restrict__ outSeq,
              float* __restrict__ outH, int T, int retSeq) {
    __shared__ float h[2 * 64];
    const int tid = threadIdx.x;
    const int b = tid >> 6, u = tid & 63;
    h[tid] = 0.f;
    __syncthreads();
    float hn = 0.f;
    for (int t = 0; t < T; ++t) {
        float rz = b1[u], rr = b1[64 + u], rh = b1[128 + u];
        const float* hb = h + b * 64;
        for (int j = 0; j < 64; ++j) {
            const float hv = hb[j];
            const float* wr = wh + j * 192;
            rz += hv * wr[u];
            rr += hv * wr[64 + u];
            rh += hv * wr[128 + u];
        }
        const float* xt = xp + ((size_t)b * T + t) * 192;
        const float z  = 1.f / (1.f + __expf(-(xt[u] + rz)));
        const float r  = 1.f / (1.f + __expf(-(xt[64 + u] + rr)));
        const float hh = tanhf(xt[128 + u] + r * rh);
        hn = z * hb[u] + (1.f - z) * hh;
        __syncthreads();
        h[tid] = hn;
        __syncthreads();
        if (retSeq) outSeq[((size_t)b * T + t) * 64 + u] = hn;
    }
    if (!retSeq) outH[tid] = hn;
}

// ---------------------------------------------------------------------------
// Dense head: d1(64->64,relu), d2(64->32,relu), throw softmax(64), tori softmax(2).
// out: [0..127] throw (2x64), [128..131] tori (2x2).
// ---------------------------------------------------------------------------
__global__ __launch_bounds__(128)
void head_kernel(const float* __restrict__ h3,
                 const float* __restrict__ d1w, const float* __restrict__ d1b,
                 const float* __restrict__ d2w, const float* __restrict__ d2b,
                 const float* __restrict__ tw,  const float* __restrict__ tb,
                 const float* __restrict__ ow,  const float* __restrict__ ob,
                 float* __restrict__ out) {
    __shared__ float s1[2 * 64], s2[2 * 64], lg[2 * 64], red[4];
    const int tid = threadIdx.x;
    const int b = tid >> 6, u = tid & 63;
    s1[tid] = h3[tid];
    __syncthreads();
    {   // d1
        float s = d1b[u];
        for (int j = 0; j < 64; ++j) s += s1[b * 64 + j] * d1w[j * 64 + u];
        s2[tid] = fmaxf(s, 0.f);
    }
    __syncthreads();
    if (u < 32) {   // d2
        float s = d2b[u];
        for (int j = 0; j < 64; ++j) s += s2[b * 64 + j] * d2w[j * 32 + u];
        s1[b * 64 + u] = fmaxf(s, 0.f);
    }
    __syncthreads();
    {   // throw logits
        float s = tb[u];
        for (int j = 0; j < 32; ++j) s += s1[b * 64 + j] * tw[j * 64 + u];
        lg[tid] = s;
    }
    __syncthreads();
    if (u == 0) {
        float m = -1e30f;
        for (int j = 0; j < 64; ++j) m = fmaxf(m, lg[b * 64 + j]);
        float sum = 0.f;
        for (int j = 0; j < 64; ++j) sum += __expf(lg[b * 64 + j] - m);
        red[b * 2 + 0] = m; red[b * 2 + 1] = sum;
    }
    __syncthreads();
    out[b * 64 + u] = __expf(lg[tid] - red[b * 2]) / red[b * 2 + 1];
    __syncthreads();
    if (u < 2) {    // tori logits
        float s = ob[u];
        for (int j = 0; j < 32; ++j) s += s1[b * 64 + j] * ow[j * 2 + u];
        lg[b * 64 + u] = s;
    }
    __syncthreads();
    if (u == 0) {
        const float m  = fmaxf(lg[b * 64], lg[b * 64 + 1]);
        const float e0 = __expf(lg[b * 64] - m), e1 = __expf(lg[b * 64 + 1] - m);
        out[128 + b * 2 + 0] = e0 / (e0 + e1);
        out[128 + b * 2 + 1] = e1 / (e0 + e1);
    }
}

// ---------------------------------------------------------------------------
// Host launcher
// ---------------------------------------------------------------------------
extern "C" void kernel_launch(void* const* d_in, const int* in_sizes, int n_in,
                              void* d_out, int out_size, void* d_ws, size_t ws_size,
                              hipStream_t stream) {
    (void)in_sizes; (void)n_in; (void)out_size; (void)ws_size;

    const float* in    = (const float*)d_in[0];
    const float* c11_k = (const float*)d_in[1];
    const float* c11_b = (const float*)d_in[2];
    const float* c12_k = (const float*)d_in[3];
    const float* c12_b = (const float*)d_in[4];
    const float* c21_k = (const float*)d_in[5];
    const float* c21_b = (const float*)d_in[6];
    const float* c22_k = (const float*)d_in[7];
    const float* c22_b = (const float*)d_in[8];
    const float* c31_k = (const float*)d_in[9];
    const float* c31_b = (const float*)d_in[10];
    const float* c32_k = (const float*)d_in[11];
    const float* c32_b = (const float*)d_in[12];
    const float* g1_wx = (const float*)d_in[13];
    const float* g1_wh = (const float*)d_in[14];
    const float* g1_b  = (const float*)d_in[15];
    const float* g2_wx = (const float*)d_in[16];
    const float* g2_wh = (const float*)d_in[17];
    const float* g2_b  = (const float*)d_in[18];
    const float* g3_wx = (const float*)d_in[19];
    const float* g3_wh = (const float*)d_in[20];
    const float* g3_b  = (const float*)d_in[21];
    const float* d1_w  = (const float*)d_in[22];
    const float* d1_b  = (const float*)d_in[23];
    const float* d2_w  = (const float*)d_in[24];
    const float* d2_b  = (const float*)d_in[25];
    const float* thr_w = (const float*)d_in[26];
    const float* thr_b = (const float*)d_in[27];
    const float* to_w  = (const float*)d_in[28];
    const float* to_b  = (const float*)d_in[29];

    // Row counts (all multiples of 64)
    const int M1 = 26 * 8 * 56 * 56;   // 652288
    const int M2 = 26 * 8 * 28 * 28;   // 163072
    const int M3 = 26 * 8 * 14 * 14;   // 40768

    char* ws = (char*)d_ws;
    size_t off = 0;
    auto alloc = [&](size_t bytes) -> char* {
        char* p = ws + off;
        off += (bytes + 255) & ~(size_t)255;
        return p;
    };
    half_t* R0 = (half_t*)alloc((size_t)M1 * 32 * 2);   // 41.75 MB ping
    half_t* R1 = (half_t*)alloc((size_t)M1 * 32 * 2);   // 41.75 MB pong
    half_t* R2 = (half_t*)alloc((size_t)M2 * 32 * 2);   // 10.44 MB pooled
    half_t* w12 = (half_t*)alloc(27648u * 2);
    half_t* w21 = (half_t*)alloc(55296u * 2);
    half_t* w22 = (half_t*)alloc(110592u * 2);
    half_t* w31 = (half_t*)alloc(221184u * 2);
    half_t* w32 = (half_t*)alloc(442368u * 2);
    float* feats = (float*)alloc(26 * 128 * 4);
    float* xpA   = (float*)alloc(26 * 192 * 4);
    float* seqA  = (float*)alloc(26 * 64 * 4);
    float* seqB  = (float*)alloc(26 * 64 * 4);
    float* h3    = (float*)alloc(128 * 4);

    // Weight conversions
    f32_to_f16<<<(27648 + 255) / 256, 256, 0, stream>>>(c12_k, w12, 27648);
    f32_to_f16<<<(55296 + 255) / 256, 256, 0, stream>>>(c21_k, w21, 55296);
    f32_to_f16<<<(110592 + 255) / 256, 256, 0, stream>>>(c22_k, w22, 110592);
    f32_to_f16<<<(221184 + 255) / 256, 256, 0, stream>>>(c31_k, w31, 221184);
    f32_to_f16<<<(442368 + 255) / 256, 256, 0, stream>>>(c32_k, w32, 442368);

    // Conv stack (NDHWC f16 activations, WMMA implicit GEMM)
    conv1_direct<<<(M1 * 32 + 255) / 256, 256, 0, stream>>>(in, c11_k, c11_b, R0);
    conv_wmma<32, 32, 56, 56><<<M1 / 64, 256, 0, stream>>>(R0, w12, c12_b, R1, M1);
    maxpool2x2<<<(M2 * 32 + 255) / 256, 256, 0, stream>>>(R1, R2, 56, 56, 32, M2 * 32);
    conv_wmma<32, 64, 28, 28><<<M2 / 64, 256, 0, stream>>>(R2, w21, c21_b, R0, M2);
    conv_wmma<64, 64, 28, 28><<<M2 / 64, 256, 0, stream>>>(R0, w22, c22_b, R1, M2);
    maxpool2x2<<<(M3 * 64 + 255) / 256, 256, 0, stream>>>(R1, R2, 28, 28, 64, M3 * 64);
    conv_wmma<64, 128, 14, 14><<<M3 / 64, 256, 0, stream>>>(R2, w31, c31_b, R0, M3);
    conv_wmma<128, 128, 14, 14><<<M3 / 64, 256, 0, stream>>>(R0, w32, c32_b, R1, M3);
    gap_kernel<<<13, 256, 0, stream>>>(R1, feats);

    // GRU stack
    gru_xp<<<(26 * 192 + 255) / 256, 256, 0, stream>>>(feats, g1_wx, g1_b, xpA, 26, 128);
    gru_scan<<<1, 128, 0, stream>>>(xpA, g1_wh, g1_b + 192, seqA, seqA, 13, 1);
    gru_xp<<<(26 * 192 + 255) / 256, 256, 0, stream>>>(seqA, g2_wx, g2_b, xpA, 26, 64);
    gru_scan<<<1, 128, 0, stream>>>(xpA, g2_wh, g2_b + 192, seqB, seqB, 13, 1);
    gru_xp<<<(26 * 192 + 255) / 256, 256, 0, stream>>>(seqB, g3_wx, g3_b, xpA, 26, 64);
    gru_scan<<<1, 128, 0, stream>>>(xpA, g3_wh, g3_b + 192, seqA, h3, 13, 0);

    // Heads
    head_kernel<<<1, 128, 0, stream>>>(h3, d1_w, d1_b, d2_w, d2_b,
                                       thr_w, thr_b, to_w, to_b, (float*)d_out);
}
